// AfmoeSDPAAttention_33371895890029
// MI455X (gfx1250) — matrix operand order repaired and checked by reading, not compiled, not run on hardware
//
#include <hip/hip_runtime.h>
#include <hip/hip_bf16.h>
#include <math.h>

typedef __attribute__((ext_vector_type(16))) _Float16 v16h;
typedef __attribute__((ext_vector_type(8)))  float    v8f;

union Frag16 { v16h h; unsigned int u[8]; };

// ---------------------------------------------------------------------------
// CDNA5 async global->LDS copy (ASYNCcnt-tracked) + wait, via inline asm.
// Each lane copies 16B from its global address to its LDS byte offset.
// ---------------------------------------------------------------------------
__device__ __forceinline__ void async_copy_b128(unsigned lds_off, const void* gaddr) {
    asm volatile("global_load_async_to_lds_b128 %0, %1, off"
                 :: "v"(lds_off), "v"(gaddr)
                 : "memory");
}
__device__ __forceinline__ void wait_async0() {
    asm volatile("s_wait_asynccnt 0x0" ::: "memory");
}

// ---------------------------------------------------------------------------
// f32 -> f16 cast
// ---------------------------------------------------------------------------
__global__ void cast_f32_f16(const float* __restrict__ x, _Float16* __restrict__ y, int n) {
    int i = blockIdx.x * blockDim.x + threadIdx.x;
    if (i < n) y[i] = (_Float16)x[i];
}

// ---------------------------------------------------------------------------
// C[M,N] = A[M,K] (f16 row-major) * W[N,K]^T (f16 row-major), f32 out.
// Block: 256 thr = 8 waves in 2(M) x 4(N); block tile 32(M) x 256(N).
//  - A tile staged in LDS via double-buffered async DMA (128-K slabs, 8KB)
//  - B fragments software-pipelined one 32-K substep ahead in registers
// ---------------------------------------------------------------------------
#define KSTEP 128

__global__ __launch_bounds__(256) void gemm_f16f32(
    const _Float16* __restrict__ A, const _Float16* __restrict__ W,
    float* __restrict__ C, int M, int N, int K)
{
    __shared__ __align__(16) _Float16 abuf[2][32 * KSTEP];

    const int tid  = threadIdx.x;
    const int wave = tid >> 5;
    const int lane = tid & 31;
    const int half = lane >> 4;
    const int l16  = lane & 15;
    const int wm   = wave >> 2;          // 0..1
    const int wn   = wave & 3;           // 0..3
    const int mblk = blockIdx.y * 32;
    const int n0   = blockIdx.x * 256 + wn * 64;

    v8f acc[4] = {};

    // B row pointers (weight rows = output columns), dword-typed
    const unsigned int* wp[4];
#pragma unroll
    for (int nn = 0; nn < 4; ++nn)
        wp[nn] = reinterpret_cast<const unsigned int*>(W + (size_t)(n0 + nn * 16 + l16) * K);

    // Async-copy mapping: buffer = 32 rows x 256B = 512 x 16B segments,
    // 2 segments per thread.
    auto issue = [&](int buf, int kb) {
#pragma unroll
        for (int i = 0; i < 2; ++i) {
            const int s   = tid + 256 * i;
            const int row = s >> 4;      // 0..31
            const int c16 = s & 15;      // 16B segment within row
            const unsigned lds = (unsigned)(uintptr_t)(&abuf[buf][0])
                               + (unsigned)(row * KSTEP * 2 + c16 * 16);
            const _Float16* g = A + (size_t)(mblk + row) * K + kb + c16 * 8;
            async_copy_b128(lds, g);
        }
    };

    // Load the 4 B fragments for 32-K substep t
    auto loadB = [&](Frag16 (&dst)[4], int t) {
#pragma unroll
        for (int nn = 0; nn < 4; ++nn)
#pragma unroll
            for (int v = 0; v < 8; ++v)
                dst[nn].u[v] = wp[nn][t * 16 + v + 8 * half];
    };

    const int tsteps = K >> 5;           // substeps of 32 K

    // One pipeline step: consume bcur (substep t), prefetch bnxt (substep t+1)
    auto step = [&](int t, Frag16 (&bcur)[4], Frag16 (&bnxt)[4]) {
        if ((t & 3) == 0) {              // new 128-K LDS slab
            wait_async0();
            __syncthreads();             // abuf[(t>>2)&1] ready for all waves
            const int kb = t * 32;
            if (kb + KSTEP < K) issue(((t >> 2) & 1) ^ 1, kb + KSTEP);
        }
        if (t + 1 < tsteps) loadB(bnxt, t + 1);   // issue ahead of WMMAs

        const unsigned int* arow = reinterpret_cast<const unsigned int*>(
            &abuf[(t >> 2) & 1][(wm * 16 + l16) * KSTEP]);
        Frag16 af;
#pragma unroll
        for (int v = 0; v < 8; ++v)      // 2x ds_load_b128
            af.u[v] = arow[(t & 3) * 16 + v + 8 * half];

#pragma unroll
        for (int nn = 0; nn < 4; ++nn)
            acc[nn] = __builtin_amdgcn_wmma_f32_16x16x32_f16(
                false, af.h, false, bcur[nn].h, (short)0, acc[nn], false, false);

        if ((t & 3) == 3) __syncthreads();   // reads of this slab done
    };

    issue(0, 0);
    Frag16 bf0[4], bf1[4];
    loadB(bf0, 0);
    for (int t = 0; t < tsteps; t += 2) {     // manual 2x unroll: static parity
        step(t,     bf0, bf1);
        step(t + 1, bf1, bf0);
    }

#pragma unroll
    for (int nn = 0; nn < 4; ++nn) {
#pragma unroll
        for (int r = 0; r < 8; ++r) {
            const int m = mblk + wm * 16 + r + 8 * half;
            const int n = n0 + nn * 16 + l16;
            C[(size_t)m * N + n] = acc[nn][r];
        }
    }
}

// ---------------------------------------------------------------------------
// Fused per-head RMS-norm + RoPE, f32 in -> f16 out. One wave per (token,head),
// lane owns d, d+32, d+64, d+96 so rotate_half is lane-local.
// ---------------------------------------------------------------------------
__global__ __launch_bounds__(32) void rmsrope_kernel(
    const float* __restrict__ X, const float* __restrict__ w,
    const float* __restrict__ cs, const float* __restrict__ sn,
    _Float16* __restrict__ out, int S, int H)
{
    const int idx  = blockIdx.x;
    const int h    = idx % H;
    const int tok  = idx / H;
    const int s    = tok % S;
    const int lane = threadIdx.x;
    const size_t base = ((size_t)tok * H + h) * 128;

    float x0 = X[base + lane],      x1 = X[base + lane + 32];
    float x2 = X[base + lane + 64], x3 = X[base + lane + 96];
    float ss = x0 * x0 + x1 * x1 + x2 * x2 + x3 * x3;
#pragma unroll
    for (int off = 16; off; off >>= 1) ss += __shfl_xor(ss, off, 32);
    const float inv = rsqrtf(ss * (1.0f / 128.0f) + 1e-6f);

    const float n0 = x0 * inv * w[lane],      n1 = x1 * inv * w[lane + 32];
    const float n2 = x2 * inv * w[lane + 64], n3 = x3 * inv * w[lane + 96];

    const float* c = cs + (size_t)s * 128;
    const float* q = sn + (size_t)s * 128;
    out[base + lane]      = (_Float16)(n0 * c[lane]      - n2 * q[lane]);
    out[base + lane + 32] = (_Float16)(n1 * c[lane + 32] - n3 * q[lane + 32]);
    out[base + lane + 64] = (_Float16)(n2 * c[lane + 64] + n0 * q[lane + 64]);
    out[base + lane + 96] = (_Float16)(n3 * c[lane + 96] + n1 * q[lane + 96]);
}

// ---------------------------------------------------------------------------
// V (b,s,hkv,d) f32  ->  Vt[b][hkv][d][s] f16  (keys contiguous for PV B-frag)
// ---------------------------------------------------------------------------
__global__ void vtrans_kernel(const float* __restrict__ V, _Float16* __restrict__ Vt,
                              int S, int total)
{
    int i = blockIdx.x * blockDim.x + threadIdx.x;
    if (i >= total) return;
    const int d   = i & 127;
    const int hv  = (i >> 7) & 3;
    const int rem = i >> 9;
    const int s   = rem % S;
    const int b   = rem / S;
    Vt[(((size_t)(b * 4 + hv) * 128 + d) * S) + s] = (_Float16)V[i];
}

// ---------------------------------------------------------------------------
// Causal flash attention, GQA 4:1. One wave per (b, h, 16-row q tile).
// 32 keys / iteration: QK^T via WMMA, online softmax in C-layout regs,
// P staged through LDS (C-layout -> A-layout), PV via WMMA.
// ---------------------------------------------------------------------------
__global__ __launch_bounds__(32) void fattn_kernel(
    const _Float16* __restrict__ Qh, const _Float16* __restrict__ Kh,
    const _Float16* __restrict__ Vt, _Float16* __restrict__ Ah, int S)
{
    const int qt   = blockIdx.x;
    const int h    = blockIdx.y;
    const int b    = blockIdx.z;
    const int hkv  = h >> 2;
    const int lane = threadIdx.x;
    const int half = lane >> 4;
    const int l16  = lane & 15;
    const float scale = 0.08838834764831845f;   // 1/sqrt(128)

    __shared__ __align__(16) _Float16 pbuf[16 * 32];

    // Q fragments for the 4 K-dim (=D) steps, A-layout
    Frag16 aq[4];
    {
        const unsigned int* qrow = reinterpret_cast<const unsigned int*>(
            Qh + (((size_t)b * S + qt * 16 + l16) * 16 + h) * 128);
#pragma unroll
        for (int ks = 0; ks < 4; ++ks)
#pragma unroll
            for (int v = 0; v < 8; ++v)
                aq[ks].u[v] = qrow[(ks * 32 + 16 * half + 2 * v) >> 1];
    }

    v8f oacc[8] = {};
    float mrow[8], lrow[8];
#pragma unroll
    for (int r = 0; r < 8; ++r) { mrow[r] = -__builtin_inff(); lrow[r] = 0.0f; }

    const int nkeys = qt * 16 + 16;
    const int nkb   = (nkeys + 31) >> 5;
    const int srow_base = qt * 16 + 8 * half;

    for (int kb = 0; kb < nkb; ++kb) {
        const int kbase = kb * 32;
        const bool act1 = (kbase + 16) < nkeys;   // wave-uniform

        v8f sa0 = {}, sa1 = {};
        {
            const unsigned int* krow = reinterpret_cast<const unsigned int*>(
                Kh + (((size_t)b * S + kbase + l16) * 4 + hkv) * 128);
#pragma unroll
            for (int ks = 0; ks < 4; ++ks) {
                Frag16 bk;
#pragma unroll
                for (int v = 0; v < 8; ++v)
                    bk.u[v] = krow[(ks * 32 + 16 * half + 2 * v) >> 1];
                sa0 = __builtin_amdgcn_wmma_f32_16x16x32_f16(
                    false, aq[ks].h, false, bk.h, (short)0, sa0, false, false);
            }
        }
        if (act1) {
            const unsigned int* krow = reinterpret_cast<const unsigned int*>(
                Kh + (((size_t)b * S + kbase + 16 + l16) * 4 + hkv) * 128);
#pragma unroll
            for (int ks = 0; ks < 4; ++ks) {
                Frag16 bk;
#pragma unroll
                for (int v = 0; v < 8; ++v)
                    bk.u[v] = krow[(ks * 32 + 16 * half + 2 * v) >> 1];
                sa1 = __builtin_amdgcn_wmma_f32_16x16x32_f16(
                    false, aq[ks].h, false, bk.h, (short)0, sa1, false, false);
            }
        }

        // online softmax over the 32-key block (C-layout: row = r+8*half, col = l16)
#pragma unroll
        for (int r = 0; r < 8; ++r) {
            const int srow = srow_base + r;
            float v0 = sa0[r] * scale;
            if (kbase + l16 > srow) v0 = -__builtin_inff();
            float v1 = act1 ? sa1[r] * scale : -__builtin_inff();
            if (kbase + 16 + l16 > srow) v1 = -__builtin_inff();

            float tm = fmaxf(v0, v1);
#pragma unroll
            for (int off = 1; off < 16; off <<= 1) tm = fmaxf(tm, __shfl_xor(tm, off, 32));
            const float mn   = fmaxf(mrow[r], tm);
            const float corr = __expf(mrow[r] - mn);
            const float p0   = __expf(v0 - mn);
            const float p1   = __expf(v1 - mn);
            float ps = p0 + p1;
#pragma unroll
            for (int off = 1; off < 16; off <<= 1) ps += __shfl_xor(ps, off, 32);
            lrow[r] = lrow[r] * corr + ps;
            mrow[r] = mn;
#pragma unroll
            for (int dt = 0; dt < 8; ++dt) oacc[dt][r] *= corr;

            pbuf[(r + 8 * half) * 32 + l16]      = (_Float16)p0;
            pbuf[(r + 8 * half) * 32 + 16 + l16] = (_Float16)p1;
        }
        __syncthreads();

        // P: C-layout -> A-layout via LDS
        Frag16 ap;
        const unsigned int* pb = reinterpret_cast<const unsigned int*>(pbuf);
#pragma unroll
        for (int v = 0; v < 8; ++v)
            ap.u[v] = pb[(l16 * 32 + 16 * half + 2 * v) >> 1];

        // PV: 8 d-tiles of 16
#pragma unroll
        for (int dt = 0; dt < 8; ++dt) {
            const unsigned int* vrow = reinterpret_cast<const unsigned int*>(
                Vt + ((size_t)(b * 4 + hkv) * 128 + dt * 16 + l16) * S + kbase);
            Frag16 bv;
#pragma unroll
            for (int v = 0; v < 8; ++v)
                bv.u[v] = vrow[v + 8 * half];
            oacc[dt] = __builtin_amdgcn_wmma_f32_16x16x32_f16(
                false, ap.h, false, bv.h, (short)0, oacc[dt], false, false);
        }
        __syncthreads();
    }

    // epilogue: normalize and store (b,s,h,d) f16
#pragma unroll
    for (int dt = 0; dt < 8; ++dt) {
#pragma unroll
        for (int r = 0; r < 8; ++r) {
            const float val = oacc[dt][r] / lrow[r];
            const int sq = qt * 16 + r + 8 * half;
            Ah[(((size_t)b * S + sq) * 16 + h) * 128 + dt * 16 + l16] = (_Float16)val;
        }
    }
}

// ---------------------------------------------------------------------------
// attn * sigmoid(gate) -> f16
// ---------------------------------------------------------------------------
__global__ void gate_kernel(const _Float16* __restrict__ Ahv, const float* __restrict__ G,
                            _Float16* __restrict__ Ag, int n)
{
    int i = blockIdx.x * blockDim.x + threadIdx.x;
    if (i < n) {
        const float a = (float)Ahv[i];
        const float g = G[i];
        Ag[i] = (_Float16)(a * (1.0f / (1.0f + __expf(-g))));
    }
}

// ---------------------------------------------------------------------------
extern "C" void kernel_launch(void* const* d_in, const int* in_sizes, int n_in,
                              void* d_out, int out_size, void* d_ws, size_t ws_size,
                              hipStream_t stream)
{
    (void)in_sizes; (void)n_in; (void)out_size; (void)ws_size;

    const float* X   = (const float*)d_in[0];
    const float* cs  = (const float*)d_in[1];
    const float* sn  = (const float*)d_in[2];
    const float* Wq  = (const float*)d_in[3];
    const float* Wk  = (const float*)d_in[4];
    const float* Wv  = (const float*)d_in[5];
    const float* Wg  = (const float*)d_in[6];
    const float* Wo  = (const float*)d_in[7];
    const float* qnw = (const float*)d_in[8];
    const float* knw = (const float*)d_in[9];
    float* out = (float*)d_out;

    const int B = 2, S = 2048, H = 2048, Hq = 16, Hkv = 4;
    const int M = B * S;                      // 4096 tokens

    char* ws = (char*)d_ws;
    size_t off = 0;
    auto alloc = [&](size_t bytes) -> void* {
        void* p = ws + off;
        off += (bytes + 255) & ~(size_t)255;
        return p;
    };

    _Float16* Xh  = (_Float16*)alloc((size_t)M * H * 2);
    _Float16* Wqh = (_Float16*)alloc((size_t)2048 * H * 2);
    _Float16* Wkh = (_Float16*)alloc((size_t)512 * H * 2);
    _Float16* Wvh = (_Float16*)alloc((size_t)512 * H * 2);
    _Float16* Wgh = (_Float16*)alloc((size_t)2048 * H * 2);
    _Float16* Woh = (_Float16*)alloc((size_t)2048 * 2048 * 2);
    float*    Qf  = (float*)alloc((size_t)M * 2048 * 4);
    float*    Kf  = (float*)alloc((size_t)M * 512 * 4);
    float*    Vf  = (float*)alloc((size_t)M * 512 * 4);
    float*    Gf  = (float*)alloc((size_t)M * 2048 * 4);
    _Float16* Qh  = (_Float16*)alloc((size_t)M * 2048 * 2);
    _Float16* Kh  = (_Float16*)alloc((size_t)M * 512 * 2);
    _Float16* Vt  = (_Float16*)alloc((size_t)M * 512 * 2);
    _Float16* Ahb = (_Float16*)Qf;   // reuse dead Qf32 region for attn f16
    _Float16* Ag  = (_Float16*)Kf;   // reuse dead Kf32+Vf32 region for gated f16

    // 1) casts
    cast_f32_f16<<<(M * H + 255) / 256, 256, 0, stream>>>(X, Xh, M * H);
    cast_f32_f16<<<(2048 * H + 255) / 256, 256, 0, stream>>>(Wq, Wqh, 2048 * H);
    cast_f32_f16<<<(512 * H + 255) / 256, 256, 0, stream>>>(Wk, Wkh, 512 * H);
    cast_f32_f16<<<(512 * H + 255) / 256, 256, 0, stream>>>(Wv, Wvh, 512 * H);
    cast_f32_f16<<<(2048 * H + 255) / 256, 256, 0, stream>>>(Wg, Wgh, 2048 * H);
    cast_f32_f16<<<(2048 * 2048 + 255) / 256, 256, 0, stream>>>(Wo, Woh, 2048 * 2048);

    // 2) projections (WMMA GEMMs, async-DMA A staging + pipelined B)
    gemm_f16f32<<<dim3(2048 / 256, M / 32), 256, 0, stream>>>(Xh, Wqh, Qf, M, 2048, H);
    gemm_f16f32<<<dim3(512 / 256,  M / 32), 256, 0, stream>>>(Xh, Wkh, Kf, M, 512,  H);
    gemm_f16f32<<<dim3(512 / 256,  M / 32), 256, 0, stream>>>(Xh, Wvh, Vf, M, 512,  H);
    gemm_f16f32<<<dim3(2048 / 256, M / 32), 256, 0, stream>>>(Xh, Wgh, Gf, M, 2048, H);

    // 3) RMS-norm + RoPE -> f16 Q/K ; V transpose -> Vt f16
    rmsrope_kernel<<<M * Hq,  32, 0, stream>>>(Qf, qnw, cs, sn, Qh, S, Hq);
    rmsrope_kernel<<<M * Hkv, 32, 0, stream>>>(Kf, knw, cs, sn, Kh, S, Hkv);
    vtrans_kernel<<<(M * 512 + 255) / 256, 256, 0, stream>>>(Vf, Vt, S, M * 512);

    // 4) causal flash attention (WMMA)
    fattn_kernel<<<dim3(S / 16, Hq, B), 32, 0, stream>>>(Qh, Kh, Vt, Ahb, S);

    // 5) sigmoid gate
    gate_kernel<<<(M * 2048 + 255) / 256, 256, 0, stream>>>(Ahb, Gf, Ag, M * 2048);

    // 6) output projection (WMMA GEMM) -> f32 out
    gemm_f16f32<<<dim3(2048 / 256, M / 32), 256, 0, stream>>>(Ag, Woh, out, M, 2048, 2048);
}
